// VectorQuantizerEMA_66305705115817
// MI455X (gfx1250) — matrix operand order repaired and checked by reading, not compiled, not run on hardware
//
#include <hip/hip_runtime.h>
#include <hip/hip_bf16.h>
#include <stdint.h>

// ---------------------------------------------------------------------------
// VectorQuantizerEMA for MI455X (gfx1250, wave32, WMMA)
// B=64 H=32 W=32 D=256 K=1024  ->  N = 65536 rows
// Distance GEMM as bf16 WMMA (v_wmma_f32_16x16x32_bf16); argmin folded into
// the epilogue via argmax(dot - 0.5*|e|^2). B tiles are staged to LDS with
// the CDNA5 async path (global_load_async_to_lds_b128 + s_wait_asynccnt),
// double-buffered so tile t+1 fetch overlaps tile t WMMAs.
// ---------------------------------------------------------------------------

#define K_CODES 1024
#define D_DIM   256
#define N_ROWS  65536
#define GAMMA_C 0.9f
#define BETA_C  0.25f
#define EPS_C   1e-5f

typedef __attribute__((ext_vector_type(16))) __bf16 v16bf;
typedef __attribute__((ext_vector_type(8)))  float  v8f;

union ABf {
  v16bf v;
  unsigned u[8];
  uint4 q[2];
};

__device__ __forceinline__ unsigned pk_bf16(float a, float b) {
  // round-to-nearest-even fp32 -> bf16, packed pair
  unsigned ua = __float_as_uint(a), ub = __float_as_uint(b);
  ua = (ua + 0x7FFFu + ((ua >> 16) & 1u)) >> 16;
  ub = (ub + 0x7FFFu + ((ub >> 16) & 1u)) >> 16;
  return (ua & 0xFFFFu) | (ub << 16);
}

// --------------------------- prep kernels ----------------------------------

// codebook [D,K] row-major -> cbT [K,D] row-major (fast quantized-row copies)
__global__ void vq_transpose(const float* __restrict__ cb, float* __restrict__ cbT) {
  int idx = blockIdx.x * blockDim.x + threadIdx.x;  // 262144 = k*256 + d
  int d = idx & 255;
  int k = idx >> 8;
  cbT[idx] = cb[d * K_CODES + k];
}

// cnhalf[k] = 0.5 * |e_k|^2 (read from contiguous cbT rows)
__global__ void vq_norm(const float* __restrict__ cbT, float* __restrict__ cnhalf) {
  int k = blockIdx.x * blockDim.x + threadIdx.x;    // 1024 threads
  const float4* r = (const float4*)(cbT + (size_t)k * D_DIM);
  float s = 0.f;
#pragma unroll 8
  for (int i = 0; i < 64; ++i) {
    float4 q = r[i];
    s += q.x * q.x + q.y * q.y + q.z * q.z + q.w * q.w;
  }
  cnhalf[k] = 0.5f * s;
}

// Pre-swizzle codebook into bf16 WMMA B-matrix lane layout.
// dword idx = (((t*8 + c)*32 + L)*8 + v)
//   t: column tile (16 codes), c: D-chunk of 32, L: lane, v: VGPR 0..7
// B 32x16 bf16 layout: lanes 0-15 hold Kdim {0..7,16..23} (2/VGPR), lanes 16-31 hold {8..15,24..31}.
__global__ void vq_prepb(const float* __restrict__ cb, unsigned* __restrict__ bp) {
  int idx = blockIdx.x * blockDim.x + threadIdx.x;  // 131072
  int v = idx & 7;
  int L = (idx >> 3) & 31;
  int c = (idx >> 8) & 7;
  int t = idx >> 11;
  int kcol = t * 16 + (L & 15);
  int d = c * 32 + (v >> 2) * 16 + (L >> 4) * 8 + 2 * (v & 3);
  bp[idx] = pk_bf16(cb[d * K_CODES + kcol], cb[(d + 1) * K_CODES + kcol]);
}

// --------------------------- main GEMM + argmax ----------------------------

__global__ void __launch_bounds__(256) vq_main(
    const float* __restrict__ inp, const uint4* __restrict__ bp4,
    const float* __restrict__ cnhalf, const float* __restrict__ cbT,
    float* __restrict__ ste, float* __restrict__ cs, float* __restrict__ dwT,
    float* __restrict__ loss_sum) {
  __shared__ uint4 lbuf[2][512];  // double-buffered 16-code x 256-d bf16 B tile

  const int tid  = threadIdx.x;
  const int lane = tid & 31;
  const int wave = tid >> 5;
  const int hh   = lane >> 4;   // lane half
  const int l16  = lane & 15;
  const int row0 = (blockIdx.x * 8 + wave) * 16;
  const int arow = row0 + l16;

  // LDS byte offset of the staging buffers (generic addr low 32 bits == LDS offset)
  const unsigned lds_base = (unsigned)(uintptr_t)(void*)&lbuf[0][0];
  const unsigned my_lds   = lds_base + (unsigned)(wave * 1024 + lane * 16);
  const char*    gbase    = (const char*)bp4 + (size_t)(wave * 1024 + lane * 16);

  // ---- Load + convert the wave's 16x256 A slab into WMMA A layout (64 VGPRs)
  // A 16x32 bf16 layout: lane holds row M=lane%16; lanes 0-15 Kdim {0..7,16..23},
  // lanes 16-31 Kdim {8..15,24..31}, 2 packed per VGPR.
  ABf a[8];
  const float* rp = inp + (size_t)arow * D_DIM;
#pragma unroll
  for (int c = 0; c < 8; ++c) {
    int db = c * 32 + hh * 8;
    float4 f0 = *(const float4*)(rp + db);
    float4 f1 = *(const float4*)(rp + db + 4);
    float4 g0 = *(const float4*)(rp + db + 16);
    float4 g1 = *(const float4*)(rp + db + 20);
    a[c].u[0] = pk_bf16(f0.x, f0.y); a[c].u[1] = pk_bf16(f0.z, f0.w);
    a[c].u[2] = pk_bf16(f1.x, f1.y); a[c].u[3] = pk_bf16(f1.z, f1.w);
    a[c].u[4] = pk_bf16(g0.x, g0.y); a[c].u[5] = pk_bf16(g0.z, g0.w);
    a[c].u[6] = pk_bf16(g1.x, g1.y); a[c].u[7] = pk_bf16(g1.z, g1.w);
  }

  float bv[8];
  int   bi[8];
#pragma unroll
  for (int i = 0; i < 8; ++i) { bv[i] = -3.4e38f; bi[i] = 0; }

  // ---- prologue: async-prefetch tile 0 into buffer 0 (each wave moves 1KB)
  {
    uint64_t src = (uint64_t)(uintptr_t)gbase;  // tile 0
    asm volatile(
        "global_load_async_to_lds_b128 %0, %1, off\n\t"
        "global_load_async_to_lds_b128 %0, %1, off offset:512"
        :: "v"(my_lds), "v"(src) : "memory");
  }

  // ---- 64 column tiles of 16 codes; D=256 reduced by 8 chained WMMAs.
  // Double-buffered: fetch of tile t+1 overlaps WMMAs of tile t.
  for (int t = 0; t < 64; ++t) {
    const int cur = t & 1;
    __syncthreads();  // everyone done READING buffer (1-cur) from iteration t-1

    {   // async-prefetch tile (t+1) mod 64 into buffer (1-cur); fire and forget
      int tn = (t + 1) & 63;
      unsigned dst = my_lds + (unsigned)((1 - cur) * 8192);
      uint64_t src = (uint64_t)(uintptr_t)(gbase + (size_t)tn * 8192);
      asm volatile(
          "global_load_async_to_lds_b128 %0, %1, off\n\t"
          "global_load_async_to_lds_b128 %0, %1, off offset:512"
          :: "v"(dst), "v"(src) : "memory");
    }
    // async loads complete in order: waiting for <=2 outstanding retires tile t's pair
    asm volatile("s_wait_asynccnt 0x2" ::: "memory");
    __syncthreads();  // all waves' tile-t writes are visible in buffer cur

    v8f acc = {0.f, 0.f, 0.f, 0.f, 0.f, 0.f, 0.f, 0.f};
#pragma unroll
    for (int c = 0; c < 8; ++c) {
      ABf b;
      b.q[0] = lbuf[cur][(c * 32 + lane) * 2];
      b.q[1] = lbuf[cur][(c * 32 + lane) * 2 + 1];
      acc = __builtin_amdgcn_wmma_f32_16x16x32_bf16(
          false, a[c].v, false, b.v, (short)0, acc, false, false);
    }

    float cn   = cnhalf[t * 16 + l16];
    int  kcand = t * 16 + l16;
#pragma unroll
    for (int i = 0; i < 8; ++i) {
      float val = acc[i] - cn;             // dot - 0.5|e|^2  (maximize)
      if (val > bv[i]) { bv[i] = val; bi[i] = kcand; }
    }
  }

  // ---- argmax reduce across the 16 lanes of each half (xor masks keep halves apart)
#pragma unroll
  for (int i = 0; i < 8; ++i) {
    float v = bv[i];
    int  id = bi[i];
#pragma unroll
    for (int m = 1; m < 16; m <<= 1) {
      float ov = __shfl_xor(v, m, 32);
      int   oi = __shfl_xor(id, m, 32);
      if (ov > v || (ov == v && oi < id)) { v = ov; id = oi; }
    }
    bv[i] = v; bi[i] = id;
  }

  // ---- epilogue: ste = quantized row, loss accumulation, EMA scatter stats
  float ll = 0.f;
#pragma unroll
  for (int i = 0; i < 8; ++i) {
    int row = row0 + hh * 8 + i;           // C layout: VGPR i -> M = i (+8 for hi half)
    int k   = bi[i];
    const float4* qr = (const float4*)(cbT + (size_t)k * D_DIM) + l16 * 4;
    const float4* xr = (const float4*)(inp + (size_t)row * D_DIM) + l16 * 4;
    float4*       sr = (float4*)(ste + (size_t)row * D_DIM) + l16 * 4;
    float*       dwp = dwT + (size_t)k * D_DIM + l16 * 16;
#pragma unroll
    for (int j = 0; j < 4; ++j) {
      float4 q = qr[j], x = xr[j];
      sr[j] = q;
      float dx = q.x - x.x, dy = q.y - x.y, dz = q.z - x.z, dwv = q.w - x.w;
      ll += dx * dx + dy * dy + dz * dz + dwv * dwv;
      atomicAdd(dwp + j * 4 + 0, x.x);
      atomicAdd(dwp + j * 4 + 1, x.y);
      atomicAdd(dwp + j * 4 + 2, x.z);
      atomicAdd(dwp + j * 4 + 3, x.w);
    }
    if (l16 == 0) atomicAdd(cs + k, 1.0f);
  }
#pragma unroll
  for (int m = 1; m < 32; m <<= 1) ll += __shfl_xor(ll, m, 32);
  if (lane == 0) atomicAdd(loss_sum, ll);
}

// --------------------------- epilogue kernels ------------------------------

__global__ void vq_stats(const float* __restrict__ cs, const float* __restrict__ ema_cs,
                         const int* __restrict__ counter, const float* __restrict__ loss_sum,
                         float* __restrict__ out, float* __restrict__ smoothed) {
  __shared__ float red[1024];
  int t = threadIdx.x;
  float bias = 1.f - powf(GAMMA_C, (float)counter[0]);
  float avg  = (ema_cs[t] * GAMMA_C + cs[t] * (1.f - GAMMA_C)) / bias;
  red[t] = avg;
  __syncthreads();
  for (int s = 512; s > 0; s >>= 1) {
    if (t < s) red[t] += red[t + s];
    __syncthreads();
  }
  float n = red[0];
  __syncthreads();
  smoothed[t] = (avg + EPS_C) / (n + (float)K_CODES * EPS_C) * n;

  float p = cs[t] / (float)N_ROWS;
  red[t]  = p * logf(p + 1e-10f);
  __syncthreads();
  for (int s = 512; s > 0; s >>= 1) {
    if (t < s) red[t] += red[t + s];
    __syncthreads();
  }
  if (t == 0) {
    out[16777216] = expf(-red[0]);                                         // perplexity
    out[16777217] = BETA_C * loss_sum[0] / ((float)N_ROWS * (float)D_DIM); // loss
  }
}

__global__ void vq_newcb(const float* __restrict__ ema_dw, const float* __restrict__ dwT,
                         const float* __restrict__ smoothed, const int* __restrict__ counter,
                         float* __restrict__ out_cb) {
  int idx = blockIdx.x * blockDim.x + threadIdx.x;  // 262144 = d*1024 + k
  int k = idx & 1023;
  int d = idx >> 10;
  float bias = 1.f - powf(GAMMA_C, (float)counter[0]);
  float avg_dw = (ema_dw[idx] * GAMMA_C + dwT[k * D_DIM + d] * (1.f - GAMMA_C)) / bias;
  out_cb[idx] = avg_dw / smoothed[k];
}

// --------------------------- launch ----------------------------------------

extern "C" void kernel_launch(void* const* d_in, const int* in_sizes, int n_in,
                              void* d_out, int out_size, void* d_ws, size_t ws_size,
                              hipStream_t stream) {
  const float* inputs   = (const float*)d_in[0];  // [64,32,32,256] f32
  const float* codebook = (const float*)d_in[1];  // [256,1024] f32
  const float* ema_cs   = (const float*)d_in[2];  // [1024] f32
  const float* ema_dw   = (const float*)d_in[3];  // [256,1024] f32
  const int*   counter  = (const int*)d_in[4];    // scalar int

  float* out = (float*)d_out;  // ste[16777216] | perplexity | loss | new_codebook[262144]

  // workspace layout (bytes); total ~2.6 MB
  char*     ws       = (char*)d_ws;
  unsigned* bp       = (unsigned*)(ws + 0);        // 524288 B  pre-swizzled bf16 B
  float*    cnhalf   = (float*)(ws + 524288);      // 4096 B    0.5*|e|^2
  float*    cbT      = (float*)(ws + 528384);      // 1048576 B codebook^T f32
  float*    cs       = (float*)(ws + 1576960);     // 4096 B    cluster sizes
  float*    dwT      = (float*)(ws + 1581056);     // 1048576 B dw^T accumulator
  float*    loss_sum = (float*)(ws + 2629632);     // 16 B
  float*    smoothed = (float*)(ws + 2629648);     // 4096 B

  // zero the accumulators (cs + dwT + loss_sum are contiguous)
  hipMemsetAsync(ws + 1576960, 0, 4096 + 1048576 + 16, stream);

  vq_transpose<<<1024, 256, 0, stream>>>(codebook, cbT);
  vq_norm<<<4, 256, 0, stream>>>(cbT, cnhalf);
  vq_prepb<<<512, 256, 0, stream>>>(codebook, bp);

  vq_main<<<512, 256, 0, stream>>>(inputs, (const uint4*)bp, cnhalf, cbT,
                                   out, cs, dwT, loss_sum);

  vq_stats<<<1, 1024, 0, stream>>>(cs, ema_cs, counter, loss_sum, out, smoothed);
  vq_newcb<<<1024, 256, 0, stream>>>(ema_dw, dwT, smoothed, counter, out + 16777218);
}
